// WindowAttention2D_81011673137635
// MI455X (gfx1250) — matrix operand order repaired and checked by reading, not compiled, not run on hardware
//
#include <hip/hip_runtime.h>

#define WHS 8
#define WWS 8
#define NTOK 64
#define NHEADS 12
#define DIM 384
#define HD 32

typedef __attribute__((ext_vector_type(16))) __bf16 v16bf;
typedef __attribute__((ext_vector_type(8)))  float  v8f;
typedef __attribute__((ext_vector_type(4)))  unsigned int v4u;

union FragBF { v16bf v; unsigned int u[8]; v4u q[2]; };

// fast float->bf16: round-to-nearest (bias+truncate). 1 add + high-half use.
__device__ __forceinline__ unsigned int fbits(float f) {
  union { float f; unsigned int u; } c; c.f = f;
  return c.u;
}
__device__ __forceinline__ unsigned short f2bf(float f) {
  return (unsigned short)((fbits(f) + 0x8000u) >> 16);
}
__device__ __forceinline__ unsigned int pack2bf(float lo, float hi) {
  unsigned int ul = fbits(lo) + 0x8000u;
  unsigned int uh = fbits(hi) + 0x8000u;
  return (ul >> 16) | (uh & 0xFFFF0000u);
}
__device__ __forceinline__ v8f wmma_bf16(v16bf a, v16bf b, v8f c) {
  return __builtin_amdgcn_wmma_f32_16x16x32_bf16(false, a, false, b, (short)0, c, false, false);
}

// ---------------- pre-pass: f32 weights -> bf16 pair words in d_ws ----------------
// Row-major [out_ch][in_ch] halfwords == WMMA B-operand order (pair index = 4*half + v, +4 when v >= 4).
#define QWRDS ((3*DIM)*(DIM/2))   // 221184 words
#define PWRDS (DIM*(DIM/2))       //  73728 words

__global__ void pack_weights(const float* __restrict__ qkv_w,
                             const float* __restrict__ proj_w,
                             unsigned int* __restrict__ wp) {
  int i = blockIdx.x * blockDim.x + threadIdx.x;
  if (i < QWRDS) {
    float2 f = ((const float2*)qkv_w)[i];
    wp[i] = pack2bf(f.x, f.y);
  } else if (i < QWRDS + PWRDS) {
    float2 f = ((const float2*)proj_w)[i - QWRDS];
    wp[i] = pack2bf(f.x, f.y);
  }
}

// ---------------- fused window attention ----------------
__global__ __launch_bounds__(384)
void swin_attn_fused(const float* __restrict__ x,
                     const unsigned int* __restrict__ wqkv,   // packed [1152][192]
                     const float* __restrict__ qkv_b,
                     const unsigned int* __restrict__ wproj,  // packed [384][192]
                     const float* __restrict__ proj_b,
                     const float* __restrict__ bias_table,
                     float* __restrict__ out) {
  extern __shared__ unsigned int smem[];
  const int lane = threadIdx.x & 31;
  const int wv   = threadIdx.x >> 5;   // wave id == head id (12 waves)
  const int hl   = lane >> 4;          // WMMA half-wave
  const int l16  = lane & 15;
  const int win  = blockIdx.x;

  unsigned int* Xp   = smem;                        // [64][192] pair words (A staging)
  unsigned int* Hreg = smem + NTOK*192 + wv*3072;   // 12 KB per head
  unsigned int* Qw   = Hreg;                        // [64][16]  row-major bf16 [tok][hd]
  unsigned int* Kw   = Hreg + 1024;                 // [64][16]  row-major bf16 [tok][hd]
  unsigned int* VTw  = Hreg + 2048;                 // [32][32]  row-major bf16 [hd][key]
  unsigned int* Pw   = Hreg;                        // [64][32]  reuses Q+K after S

  const v8f vzero = {0.f,0.f,0.f,0.f,0.f,0.f,0.f,0.f};

  // ---------- Stage 1: x window -> LDS bf16 ----------
  {
    const float2* xw = (const float2*)(x + (size_t)win * (NTOK * DIM));
    #pragma unroll
    for (int t = 0; t < 32; ++t) {
      int idx = (int)threadIdx.x + t * 384;   // word index = tok*192 + pair
      float2 f = xw[idx];
      Xp[idx] = pack2bf(f.x, f.y);
    }
  }
  __syncthreads();

  // ---------- Stage 2: per-head QKV GEMM ----------
  #pragma unroll 1
  for (int s = 0; s < 3; ++s) {
    const int ob = s * DIM + wv * HD;
    v8f acc[4][2];
    #pragma unroll
    for (int mi = 0; mi < 4; ++mi)
      #pragma unroll
      for (int ni = 0; ni < 2; ++ni) acc[mi][ni] = vzero;

    #pragma unroll 1
    for (int kstep = 0; kstep < 12; ++kstep) {
      FragBF a[4];
      #pragma unroll
      for (int mi = 0; mi < 4; ++mi) {
        const unsigned int* row = Xp + (16*mi + l16) * 192 + kstep * 16 + 4*hl;
        a[mi].q[0] = *(const v4u*)(row);
        a[mi].q[1] = *(const v4u*)(row + 8);
      }
      FragBF b[2];
      #pragma unroll
      for (int ni = 0; ni < 2; ++ni) {
        const unsigned int* wr = wqkv + (size_t)(ob + ni*16 + l16) * 192 + kstep * 16 + 4*hl;
        b[ni].q[0] = *(const v4u*)(wr);
        b[ni].q[1] = *(const v4u*)(wr + 8);
      }
      #pragma unroll
      for (int mi = 0; mi < 4; ++mi)
        #pragma unroll
        for (int ni = 0; ni < 2; ++ni)
          acc[mi][ni] = wmma_bf16(a[mi].v, b[ni].v, acc[mi][ni]);
    }

    float bb[2];
    #pragma unroll
    for (int ni = 0; ni < 2; ++ni) bb[ni] = qkv_b[ob + ni*16 + l16];

    if (s == 2) {                 // V: transposed [hd][key]; key-pairs live in-lane -> dword stores
      #pragma unroll
      for (int mi = 0; mi < 4; ++mi)
        #pragma unroll
        for (int ni = 0; ni < 2; ++ni) {
          int hd = 16*ni + l16;
          #pragma unroll
          for (int r = 0; r < 8; r += 2) {
            unsigned int w = pack2bf(acc[mi][ni][r] + bb[ni], acc[mi][ni][r+1] + bb[ni]);
            VTw[hd * (NTOK/2) + 8*mi + 4*hl + (r >> 1)] = w;
          }
        }
    } else {                      // Q and K: row-major [tok][hd], 16-bit stores
      unsigned short* Dh = (unsigned short*)(s == 0 ? Qw : Kw);
      #pragma unroll
      for (int mi = 0; mi < 4; ++mi)
        #pragma unroll
        for (int ni = 0; ni < 2; ++ni)
          #pragma unroll
          for (int r = 0; r < 8; ++r) {
            int tok = 16*mi + 8*hl + r, hd = 16*ni + l16;
            Dh[tok * HD + hd] = f2bf(acc[mi][ni][r] + bb[ni]);
          }
    }
  }
  __syncthreads();   // all waves done reading Xp (X consumed)

  // ---------- Stage 3: S = Q K^T, softmax, O = P V ----------
  v8f sacc[4][4];
  #pragma unroll
  for (int mi = 0; mi < 4; ++mi)
    #pragma unroll
    for (int ni = 0; ni < 4; ++ni) sacc[mi][ni] = vzero;
  {
    FragBF qa[4];
    #pragma unroll
    for (int mi = 0; mi < 4; ++mi) {
      const unsigned int* row = Qw + (16*mi + l16) * 16 + 4*hl;
      qa[mi].q[0] = *(const v4u*)(row);
      qa[mi].q[1] = *(const v4u*)(row + 8);
    }
    FragBF kb[4];
    #pragma unroll
    for (int ni = 0; ni < 4; ++ni) {
      const unsigned int* row = Kw + (16*ni + l16) * 16 + 4*hl;
      kb[ni].q[0] = *(const v4u*)(row);
      kb[ni].q[1] = *(const v4u*)(row + 8);
    }
    #pragma unroll
    for (int mi = 0; mi < 4; ++mi)
      #pragma unroll
      for (int ni = 0; ni < 4; ++ni)
        sacc[mi][ni] = wmma_bf16(qa[mi].v, kb[ni].v, sacc[mi][ni]);
  }

  // scale + relative-position bias + row softmax (rows live in 16-lane halves)
  {
    const float scale = 0.17677669529663689f;  // 32^-0.5
    unsigned short* Ph = (unsigned short*)Pw;
    #pragma unroll
    for (int mi = 0; mi < 4; ++mi) {
      #pragma unroll
      for (int r = 0; r < 8; ++r) {
        int qt = 16*mi + 8*hl + r;
        int qi = qt >> 3, qj = qt & 7;
        float rv[4];
        #pragma unroll
        for (int ni = 0; ni < 4; ++ni) {
          int kt = 16*ni + l16;
          int ridx = (qi - (kt >> 3) + (WHS-1)) * (2*WWS-1) + (qj - (kt & 7) + (WWS-1));
          rv[ni] = sacc[mi][ni][r] * scale + bias_table[ridx * NHEADS + wv];
        }
        float mx = fmaxf(fmaxf(rv[0], rv[1]), fmaxf(rv[2], rv[3]));
        #pragma unroll
        for (int off = 1; off < 16; off <<= 1)
          mx = fmaxf(mx, __shfl_xor(mx, off, 32));
        float sum = 0.f;
        #pragma unroll
        for (int ni = 0; ni < 4; ++ni) { rv[ni] = __expf(rv[ni] - mx); sum += rv[ni]; }
        #pragma unroll
        for (int off = 1; off < 16; off <<= 1)
          sum += __shfl_xor(sum, off, 32);
        float inv = 1.0f / sum;
        #pragma unroll
        for (int ni = 0; ni < 4; ++ni)
          Ph[qt * NTOK + 16*ni + l16] = f2bf(rv[ni] * inv);
      }
    }
  }

  // O = P @ V   ([64x64] @ [64x32], 2 k-steps)
  v8f oacc[4][2];
  #pragma unroll
  for (int mi = 0; mi < 4; ++mi)
    #pragma unroll
    for (int ni = 0; ni < 2; ++ni) oacc[mi][ni] = vzero;
  #pragma unroll
  for (int ks = 0; ks < 2; ++ks) {
    FragBF pa[4];
    #pragma unroll
    for (int mi = 0; mi < 4; ++mi) {
      const unsigned int* row = Pw + (16*mi + l16) * 32 + ks * 16 + 4*hl;
      pa[mi].q[0] = *(const v4u*)(row);
      pa[mi].q[1] = *(const v4u*)(row + 8);
    }
    FragBF vb[2];
    #pragma unroll
    for (int ni = 0; ni < 2; ++ni) {
      const unsigned int* row = VTw + (16*ni + l16) * 32 + ks * 16 + 4*hl;
      vb[ni].q[0] = *(const v4u*)(row);
      vb[ni].q[1] = *(const v4u*)(row + 8);
    }
    #pragma unroll
    for (int mi = 0; mi < 4; ++mi)
      #pragma unroll
      for (int ni = 0; ni < 2; ++ni)
        oacc[mi][ni] = wmma_bf16(pa[mi].v, vb[ni].v, oacc[mi][ni]);
  }

  // head outputs -> Xp (A-staging for projection), bf16 row-major [tok][384]
  {
    unsigned short* Xh = (unsigned short*)Xp;
    #pragma unroll
    for (int mi = 0; mi < 4; ++mi)
      #pragma unroll
      for (int ni = 0; ni < 2; ++ni)
        #pragma unroll
        for (int r = 0; r < 8; ++r) {
          int tok = 16*mi + 8*hl + r;
          int ch  = wv*HD + 16*ni + l16;
          Xh[tok*DIM + ch] = f2bf(oacc[mi][ni][r]);
        }
  }
  __syncthreads();

  // ---------- Stage 4: projection ----------
  {
    const int ob = wv * 32;
    v8f acc[4][2];
    #pragma unroll
    for (int mi = 0; mi < 4; ++mi)
      #pragma unroll
      for (int ni = 0; ni < 2; ++ni) acc[mi][ni] = vzero;

    #pragma unroll 1
    for (int kstep = 0; kstep < 12; ++kstep) {
      FragBF a[4];
      #pragma unroll
      for (int mi = 0; mi < 4; ++mi) {
        const unsigned int* row = Xp + (16*mi + l16) * 192 + kstep * 16 + 4*hl;
        a[mi].q[0] = *(const v4u*)(row);
        a[mi].q[1] = *(const v4u*)(row + 8);
      }
      FragBF b[2];
      #pragma unroll
      for (int ni = 0; ni < 2; ++ni) {
        const unsigned int* wr = wproj + (size_t)(ob + ni*16 + l16) * 192 + kstep * 16 + 4*hl;
        b[ni].q[0] = *(const v4u*)(wr);
        b[ni].q[1] = *(const v4u*)(wr + 8);
      }
      #pragma unroll
      for (int mi = 0; mi < 4; ++mi)
        #pragma unroll
        for (int ni = 0; ni < 2; ++ni)
          acc[mi][ni] = wmma_bf16(a[mi].v, b[ni].v, acc[mi][ni]);
    }

    float pb[2];
    #pragma unroll
    for (int ni = 0; ni < 2; ++ni) pb[ni] = proj_b[ob + 16*ni + l16];
    float* ow = out + (size_t)win * (NTOK * DIM);
    #pragma unroll
    for (int mi = 0; mi < 4; ++mi)
      #pragma unroll
      for (int ni = 0; ni < 2; ++ni)
        #pragma unroll
        for (int r = 0; r < 8; ++r) {
          int tok = 16*mi + 8*hl + r;
          ow[tok*DIM + ob + 16*ni + l16] = acc[mi][ni][r] + pb[ni];
        }
  }
}

extern "C" void kernel_launch(void* const* d_in, const int* in_sizes, int n_in,
                              void* d_out, int out_size, void* d_ws, size_t ws_size,
                              hipStream_t stream) {
  const float* x          = (const float*)d_in[0];
  const float* qkv_w      = (const float*)d_in[1];
  const float* qkv_b      = (const float*)d_in[2];
  const float* proj_w     = (const float*)d_in[3];
  const float* proj_b     = (const float*)d_in[4];
  const float* bias_table = (const float*)d_in[5];
  float* out = (float*)d_out;

  unsigned int* wp = (unsigned int*)d_ws;               // 1.18 MB packed weights
  const int packN = QWRDS + PWRDS;
  pack_weights<<<(packN + 255) / 256, 256, 0, stream>>>(qkv_w, proj_w, wp);

  const int Bn = in_sizes[0] / (NTOK * DIM);            // 4096 windows
  const size_t shmem = (size_t)(NTOK*192 + NHEADS*3072) * sizeof(unsigned int); // 192 KB

  swin_attn_fused<<<Bn, NHEADS * 32, shmem, stream>>>(
      x, wp, qkv_b, wp + QWRDS, proj_b, bias_table, out);
}
